// VQFRv1_83124797047605
// MI455X (gfx1250) — compile-verified, matrix operand order checked by reading, not run on hardware
//
#include <hip/hip_runtime.h>
#include <hip/hip_bf16.h>

typedef __attribute__((ext_vector_type(16))) _Float16 v16h;
typedef __attribute__((ext_vector_type(8)))  _Float16 v8h;
typedef __attribute__((ext_vector_type(8)))  float    v8f;

#define BN 8
#define CHN 256
#define HH 32
#define WW 32
#define HW 1024
#define NPOS 8192
#define NCODE 1024
#define NDG 4
#define CG 64

__device__ __forceinline__ v8f wmma16(const v16h& a, const v16h& b, const v8f& c) {
    // D = A(16x32 f16) x B(32x16 f16) + C(16x16 f32)
    return __builtin_amdgcn_wmma_f32_16x16x32_f16(false, a, false, b, (short)0, c, false, false);
}

// A/B fragment K index for element j, per CDNA5 16-bit 16x32 layout:
// K = (j&7) + 16*(j>=8) + 8*(lane>=16)  -> two contiguous 8-half runs.
__device__ __forceinline__ int kmap(int lane, int j) {
    return (j & 7) + ((j >> 3) << 4) + ((lane >> 4) << 3);
}

// Build a fragment from a 32-half row (f16, 16B-aligned) for half `hi` (= lane>>4):
// two 16-byte loads at half-offsets {8*hi, 16+8*hi}.
__device__ __forceinline__ v16h frag_row(const _Float16* row, int hi) {
    const v8h* p = (const v8h*)row;
    v8h lo = p[hi];
    v8h hh = p[2 + hi];
    return __builtin_shufflevector(lo, hh, 0, 1, 2, 3, 4, 5, 6, 7,
                                   8, 9, 10, 11, 12, 13, 14, 15);
}

// ---------------- prep kernels ----------------

__global__ void k_init_loss(float* loss) { if (threadIdx.x == 0) loss[0] = 0.f; }

// f32 [Cout][Ktot] -> f16 [Cout][Kpad], zero-padded (Kpad % 32 == 0)
__global__ void k_prep_w(const float* __restrict__ w, _Float16* __restrict__ wh,
                         int Cout, int Ktot, int Kpad) {
    int t = blockIdx.x * 256 + threadIdx.x;
    if (t < Cout * Kpad) {
        int co = t / Kpad, k = t - co * Kpad;
        wh[t] = (k < Ktot) ? (_Float16)w[(size_t)co * Ktot + k] : (_Float16)0.f;
    }
}

__global__ void k_codenorm(const float* __restrict__ cb, float* __restrict__ cn) {
    int i = blockIdx.x * blockDim.x + threadIdx.x;
    if (i < NCODE) {
        const float* r = cb + (size_t)i * CHN;
        float s = 0.f;
        for (int c = 0; c < CHN; ++c) { float v = r[c]; s += v * v; }
        cn[i] = s;
    }
}

// ---------------- VQ argmin via WMMA (z . c^T) ----------------
// block = 1 wave (32), grid.x = 512 position tiles of 16
__global__ void k_vq_argmin(const float* __restrict__ ze, const _Float16* __restrict__ cbh,
                            const float* __restrict__ cn, int* __restrict__ idx) {
    const int lane = threadIdx.x & 31;
    const int p0 = blockIdx.x * 16;
    const int b = p0 / HW;
    const int hw0 = p0 % HW;
    const int n = lane & 15;
    const int hi = lane >> 4;
    const int hw = hw0 + n;
    const float* zbase = ze + (size_t)b * CHN * HW + hw;   // + k*HW

    v16h afr[8];
    for (int ks = 0; ks < 8; ++ks)
        for (int j = 0; j < 16; ++j) {
            int k = ks * 32 + kmap(lane, j);
            afr[ks][j] = (_Float16)zbase[(size_t)k * HW];
        }

    float bestd[8]; int besti[8];
    for (int r = 0; r < 8; ++r) { bestd[r] = 3.4e38f; besti[r] = 0; }

    for (int nt = 0; nt < NCODE / 16; ++nt) {
        int code = nt * 16 + n;
        const _Float16* crow = cbh + (size_t)code * CHN;
        v8f acc = {};
        for (int ks = 0; ks < 8; ++ks) {
            v16h bf = frag_row(crow + ks * 32, hi);
            acc = wmma16(afr[ks], bf, acc);
        }
        float cnv = cn[code];
        for (int r = 0; r < 8; ++r) {
            float d = cnv - 2.f * acc[r];           // |z|^2 constant per row: dropped
            bool t = (d < bestd[r]) || (d == bestd[r] && code < besti[r]);
            bestd[r] = t ? d : bestd[r];
            besti[r] = t ? code : besti[r];
        }
    }
    for (int off = 1; off < 16; off <<= 1) {
        for (int r = 0; r < 8; ++r) {
            float od = __shfl_xor(bestd[r], off, 32);
            int   oi = __shfl_xor(besti[r], off, 32);
            bool t = (od < bestd[r]) || (od == bestd[r] && oi < besti[r]);
            bestd[r] = t ? od : bestd[r];
            besti[r] = t ? oi : besti[r];
        }
    }
    if (n == 0) {
        for (int r = 0; r < 8; ++r) idx[p0 + r + hi * 8] = besti[r];
    }
}

// hq = codebook[idx] (NCHW); accumulate sum((zq - z_e)^2)
__global__ void k_gather_hq_loss(const float* __restrict__ ze, const float* __restrict__ cb,
                                 const int* __restrict__ idx, float* __restrict__ hq,
                                 float* __restrict__ loss) {
    __shared__ float red[256];
    int t = blockIdx.x * 256 + threadIdx.x;   // over CHN*NPOS
    int c = t / NPOS;
    int p = t % NPOS;
    int b = p / HW, hw = p % HW;
    float zq = cb[(size_t)idx[p] * CHN + c];
    size_t oi = (size_t)b * CHN * HW + (size_t)c * HW + hw;
    float diff = zq - ze[oi];
    hq[oi] = zq;
    red[threadIdx.x] = diff * diff;
    __syncthreads();
    for (int s = 128; s > 0; s >>= 1) {
        if (threadIdx.x < s) red[threadIdx.x] += red[threadIdx.x + s];
        __syncthreads();
    }
    if (threadIdx.x == 0) atomicAdd(loss, red[0]);
}

// ---------------- implicit-im2col conv as WMMA GEMM ----------------
// block = 128 (4 waves); M-tile = 32 positions (2 subtiles), each wave 64 couts.
// Double-buffered LDS A-tile: one barrier per K-step, staging overlaps WMMA.
template<int KSZ>
__global__ void k_conv_wmma(const float* __restrict__ in0, int C0,
                            const float* __restrict__ in1, int C1,
                            const _Float16* __restrict__ wh, const float* __restrict__ bias,
                            const float* __restrict__ addsrc,
                            float* __restrict__ out, int Cout, int pad, int Kpad) {
    __shared__ _Float16 lA[2][32][32];
    const int C0k = C0 * KSZ * KSZ;
    const int Ktot = (C0 + C1) * KSZ * KSZ;
    const int lane = threadIdx.x & 31;
    const int wave = threadIdx.x >> 5;
    const int p0 = blockIdx.x * 32;
    const int b = p0 / HW;
    const int hw0 = p0 % HW;
    const int n = lane & 15;
    const int hi = lane >> 4;
    const int co_base = wave * 64;
    v8f acc[2][4] = {};
    const int nsteps = Kpad >> 5;

    auto stage = [&](int bufi, int ks) {
        for (int e = threadIdx.x; e < 1024; e += 128) {
            int i = e >> 5, kl = e & 31;
            int kk = ks + kl;
            float v = 0.f;
            if (kk < Ktot) {
                bool first = (kk < C0k);
                int kr = first ? kk : (kk - C0k);
                int ci = kr / (KSZ * KSZ);
                int tt = kr - ci * (KSZ * KSZ);
                int ky = tt / KSZ, kx = tt - (tt / KSZ) * KSZ;
                int hw = hw0 + i;
                int y = (hw >> 5) + ky - pad;       // W == 32
                int x = (hw & 31) + kx - pad;
                if (y >= 0 && y < HH && x >= 0 && x < WW) {
                    v = first ? in0[(size_t)b * C0 * HW + (size_t)ci * HW + y * WW + x]
                              : in1[(size_t)b * C1 * HW + (size_t)ci * HW + y * WW + x];
                }
            }
            lA[bufi][i][kl] = (_Float16)v;
        }
    };

    stage(0, 0);
    for (int st = 0; st < nsteps; ++st) {
        __syncthreads();
        if (st + 1 < nsteps) stage((st + 1) & 1, (st + 1) * 32);
        const int ks = st * 32;
        v16h a0 = frag_row(&lA[st & 1][n][0], hi);
        v16h a1 = frag_row(&lA[st & 1][16 + n][0], hi);
        for (int t = 0; t < 4; ++t) {
            int co = co_base + t * 16 + n;
            v16h bf;
            if (co < Cout) bf = frag_row(wh + (size_t)co * Kpad + ks, hi);
            else           bf = (v16h){};
            acc[0][t] = wmma16(a0, bf, acc[0][t]);
            acc[1][t] = wmma16(a1, bf, acc[1][t]);
        }
    }

    for (int u = 0; u < 2; ++u) {
        for (int t = 0; t < 4; ++t) {
            int co = co_base + t * 16 + n;
            if (co < Cout) {
                float bi = bias[co];
                for (int r = 0; r < 8; ++r) {
                    int m = u * 16 + r + (hi << 3);
                    size_t oi = (size_t)b * Cout * HW + (size_t)co * HW + (hw0 + m);
                    float v = acc[u][t][r] + bi;
                    if (addsrc) v += addsrc[oi];
                    out[oi] = v;
                }
            }
        }
    }
}

// ---------------- GroupNorm (+ optional SiLU), concat-capable input ----------------
__global__ void k_gn_silu(const float* __restrict__ in0, int C0,
                          const float* __restrict__ in1, int C1,
                          const float* __restrict__ g, const float* __restrict__ be,
                          float* __restrict__ out, int dosilu) {
    __shared__ float s1[256], s2[256];
    const int Cn = C0 + C1;
    const int cpg = Cn / 32;
    const int b = blockIdx.x >> 5;
    const int grp = blockIdx.x & 31;
    const int n = cpg * HW;
    float sum = 0.f, sq = 0.f;
    for (int e = threadIdx.x; e < n; e += 256) {
        int c = grp * cpg + e / HW;
        int hw = e % HW;
        float v = (c < C0) ? in0[(size_t)b * C0 * HW + (size_t)c * HW + hw]
                           : in1[(size_t)b * C1 * HW + (size_t)(c - C0) * HW + hw];
        sum += v; sq += v * v;
    }
    s1[threadIdx.x] = sum; s2[threadIdx.x] = sq;
    __syncthreads();
    for (int s = 128; s > 0; s >>= 1) {
        if (threadIdx.x < s) { s1[threadIdx.x] += s1[threadIdx.x + s]; s2[threadIdx.x] += s2[threadIdx.x + s]; }
        __syncthreads();
    }
    float mu = s1[0] / (float)n;
    float var = s2[0] / (float)n - mu * mu;
    float rs = rsqrtf(var + 1e-6f);
    for (int e = threadIdx.x; e < n; e += 256) {
        int c = grp * cpg + e / HW;
        int hw = e % HW;
        float v = (c < C0) ? in0[(size_t)b * C0 * HW + (size_t)c * HW + hw]
                           : in1[(size_t)b * C1 * HW + (size_t)(c - C0) * HW + hw];
        float y = (v - mu) * rs * g[c] + be[c];
        if (dosilu) y = y / (1.f + __expf(-y));
        out[(size_t)b * Cn * HW + (size_t)c * HW + hw] = y;
    }
}

// ---------------- depthwise 7x7 pad 3 ----------------
__global__ void k_dw7(const float* __restrict__ in, const float* __restrict__ w,
                      const float* __restrict__ bias, float* __restrict__ out) {
    int t = blockIdx.x * 256 + threadIdx.x;     // B*C*HW
    int hw = t % HW;
    int bc = t / HW;
    int c = bc % CHN;
    int y = hw >> 5, x = hw & 31;
    const float* ip = in + (size_t)bc * HW;
    const float* wp = w + (size_t)c * 49;
    float s = bias[c];
    for (int ky = 0; ky < 7; ++ky) {
        int yy = y + ky - 3;
        if (yy < 0 || yy >= HH) continue;
        for (int kx = 0; kx < 7; ++kx) {
            int xx = x + kx - 3;
            if (xx < 0 || xx >= WW) continue;
            s += wp[ky * 7 + kx] * ip[yy * WW + xx];
        }
    }
    out[t] = s;
}

// ---------------- deformable conv 3x3 via WMMA ----------------
// block = 128 (4 waves x 4 N-tiles = 256 couts), M-tile = 32, double-buffered A.
__global__ void k_deform_wmma(const float* __restrict__ hq, const float* __restrict__ om,
                              const _Float16* __restrict__ wh, const float* __restrict__ bias,
                              float* __restrict__ out) {
    __shared__ _Float16 lA[2][32][32];
    __shared__ float sY0[9][32], sX0[9][32], sWY[9][32], sWX[9][32], sM[9][32];
    const int lane = threadIdx.x & 31;
    const int wave = threadIdx.x >> 5;
    const int p0 = blockIdx.x * 32;
    const int b = p0 / HW;
    const int hw0 = p0 % HW;
    const int n = lane & 15;
    const int hi = lane >> 4;
    const int co_base = wave * 64;
    v8f acc[2][4] = {};

    auto stage = [&](int bufi, int gidx, int ks) {
        for (int e = threadIdx.x; e < 1024; e += 128) {
            int i = e >> 5, kl = e & 31;
            int kk = ks + kl;                   // kk = c*9 + k, < 576
            int c = kk / 9, k = kk - (kk / 9) * 9;
            int y0 = (int)sY0[k][i], x0 = (int)sX0[k][i];
            float wy = sWY[k][i], wx = sWX[k][i];
            const float* ip = hq + (size_t)b * CHN * HW + (size_t)(gidx * CG + c) * HW;
            float v00 = (y0 >= 0 && y0 < HH && x0 >= 0 && x0 < WW) ? ip[y0 * WW + x0] : 0.f;
            float v01 = (y0 >= 0 && y0 < HH && x0 + 1 >= 0 && x0 + 1 < WW) ? ip[y0 * WW + x0 + 1] : 0.f;
            float v10 = (y0 + 1 >= 0 && y0 + 1 < HH && x0 >= 0 && x0 < WW) ? ip[(y0 + 1) * WW + x0] : 0.f;
            float v11 = (y0 + 1 >= 0 && y0 + 1 < HH && x0 + 1 >= 0 && x0 + 1 < WW) ? ip[(y0 + 1) * WW + x0 + 1] : 0.f;
            float v = (v00 * (1.f - wy) * (1.f - wx) + v01 * (1.f - wy) * wx +
                       v10 * wy * (1.f - wx) + v11 * wy * wx) * sM[k][i];
            lA[bufi][i][kl] = (_Float16)v;
        }
    };

    for (int gidx = 0; gidx < NDG; ++gidx) {
        __syncthreads();
        // stage bilinear sampling params: 9 taps x 32 positions
        for (int e = threadIdx.x; e < 288; e += 128) {
            int k = e / 32, i = e % 32;
            int hw = hw0 + i;
            int y = hw >> 5, x = hw & 31;
            size_t ob = (size_t)b * 108 * HW + hw;
            float dy = om[ob + (size_t)(gidx * 9 + k) * HW];
            float dx = om[ob + (size_t)(36 + gidx * 9 + k) * HW];
            float mr = om[ob + (size_t)(72 + gidx * 9 + k) * HW];
            float py = (float)y + (float)(k / 3) - 1.f + dy;
            float px = (float)x + (float)(k % 3) - 1.f + dx;
            float y0f = floorf(py), x0f = floorf(px);
            sY0[k][i] = y0f; sX0[k][i] = x0f;
            sWY[k][i] = py - y0f; sWX[k][i] = px - x0f;
            sM[k][i] = 1.f / (1.f + __expf(-mr));
        }
        __syncthreads();
        stage(0, gidx, 0);
        for (int st = 0; st < 18; ++st) {           // group K = 64*9 = 576
            __syncthreads();
            if (st + 1 < 18) stage((st + 1) & 1, gidx, (st + 1) * 32);
            const int ks = st * 32;
            v16h a0 = frag_row(&lA[st & 1][n][0], hi);
            v16h a1 = frag_row(&lA[st & 1][16 + n][0], hi);
            for (int t = 0; t < 4; ++t) {
                int co = co_base + t * 16 + n;
                // wh row-major [co][2304]; group g occupies kk = g*576 + (c*9+k)
                v16h bf = frag_row(wh + (size_t)co * 2304 + gidx * 576 + ks, hi);
                acc[0][t] = wmma16(a0, bf, acc[0][t]);
                acc[1][t] = wmma16(a1, bf, acc[1][t]);
            }
        }
    }
    for (int u = 0; u < 2; ++u) {
        for (int t = 0; t < 4; ++t) {
            int co = co_base + t * 16 + n;
            float bi = bias[co];
            for (int r = 0; r < 8; ++r) {
                int m = u * 16 + r + (hi << 3);
                out[(size_t)b * CHN * HW + (size_t)co * HW + (hw0 + m)] = acc[u][t][r] + bi;
            }
        }
    }
}

__global__ void k_finalize_loss(const float* __restrict__ loss, float* __restrict__ out) {
    if (threadIdx.x == 0) out[0] = 1.25f * loss[0] / (float)(BN * CHN * HW);
}

// ---------------- host side ----------------

extern "C" void kernel_launch(void* const* d_in, const int* in_sizes, int n_in,
                              void* d_out, int out_size, void* d_ws, size_t ws_size,
                              hipStream_t stream) {
    (void)in_sizes; (void)n_in; (void)out_size; (void)ws_size;
    const float* ze   = (const float*)d_in[0];
    const float* cb   = (const float*)d_in[1];
    const float* prior= (const float*)d_in[2];
    const float* xlq  = (const float*)d_in[3];
    const float* oc1_c1_w = (const float*)d_in[4];
    const float* oc1_c1_b = (const float*)d_in[5];
    const float* oc1_gn1_g= (const float*)d_in[6];
    const float* oc1_gn1_b= (const float*)d_in[7];
    const float* oc1_dw_w = (const float*)d_in[8];
    const float* oc1_dw_b = (const float*)d_in[9];
    const float* oc1_gn2_g= (const float*)d_in[10];
    const float* oc1_gn2_b= (const float*)d_in[11];
    const float* oc1_c2_w = (const float*)d_in[12];
    const float* oc1_c2_b = (const float*)d_in[13];
    const float* oc2_w    = (const float*)d_in[14];
    const float* oc2_b    = (const float*)d_in[15];
    const float* oc2_gn_g = (const float*)d_in[16];
    const float* oc2_gn_b = (const float*)d_in[17];
    const float* off_w    = (const float*)d_in[18];
    const float* off_b    = (const float*)d_in[19];
    const float* dcn_w    = (const float*)d_in[20];
    const float* dcn_b    = (const float*)d_in[21];
    const float* rb_gn1_g = (const float*)d_in[22];
    const float* rb_gn1_b = (const float*)d_in[23];
    const float* rb_c1_w  = (const float*)d_in[24];
    const float* rb_c1_b  = (const float*)d_in[25];
    const float* rb_gn2_g = (const float*)d_in[26];
    const float* rb_gn2_b = (const float*)d_in[27];
    const float* rb_c2_w  = (const float*)d_in[28];
    const float* rb_c2_b  = (const float*)d_in[29];
    const float* rb_sc_w  = (const float*)d_in[30];
    const float* rb_sc_b  = (const float*)d_in[31];
    const float* out_gn_g = (const float*)d_in[32];
    const float* out_gn_b = (const float*)d_in[33];
    const float* out_w    = (const float*)d_in[34];
    const float* out_b    = (const float*)d_in[35];

    float* ws = (float*)d_ws;
    const size_t ACT = (size_t)BN * CHN * HW;          // 2,097,152
    float* cn   = ws;                                  // 1024
    float* loss = ws + 1024;                           // 1
    int*   idx  = (int*)(ws + 2048);                   // 8192 ints
    float* hq   = ws + 16384;
    float* t0   = hq + ACT;
    float* t1   = t0 + ACT;
    float* om   = t1 + ACT;                            // 8*108*1024
    float* warp = om + (size_t)BN * 108 * HW;
    float* gnb  = warp + ACT;                          // 512-channel buffer
    float* xb   = gnb + 2 * ACT;

    // f16 weight pool (16B-aligned; all sizes multiple of 8 halves)
    _Float16* hp = (_Float16*)(xb + ACT);
    _Float16* cbh     = hp;               hp += (size_t)NCODE * CHN;
    _Float16* wh_c1   = hp;               hp += (size_t)256 * 512;
    _Float16* wh_c2   = hp;               hp += (size_t)256 * 256;
    _Float16* wh_oc2  = hp;               hp += (size_t)256 * 2304;
    _Float16* wh_off  = hp;               hp += (size_t)108 * 2304;
    _Float16* wh_dcn  = hp;               hp += (size_t)256 * 2304;
    _Float16* wh_rc1  = hp;               hp += (size_t)256 * 4608;
    _Float16* wh_rsc  = hp;               hp += (size_t)256 * 512;
    _Float16* wh_rc2  = hp;               hp += (size_t)256 * 2304;
    _Float16* wh_out  = hp;               hp += (size_t)3 * 2336;      // Ktot=2331 padded

    float* img = (float*)d_out;                        // 8*3*32*32
    float* lossout = img + (size_t)BN * 3 * HW;

    dim3 cblk(128);
    auto blocks = [](size_t e) { return (unsigned)((e + 255) / 256); };

    k_init_loss<<<1, 32, 0, stream>>>(loss);
    k_codenorm<<<4, 256, 0, stream>>>(cb, cn);
    k_prep_w<<<blocks(262144), 256, 0, stream>>>(cb, cbh, NCODE, CHN, CHN);
    k_prep_w<<<blocks(256 * 512), 256, 0, stream>>>(oc1_c1_w, wh_c1, 256, 512, 512);
    k_prep_w<<<blocks(256 * 256), 256, 0, stream>>>(oc1_c2_w, wh_c2, 256, 256, 256);
    k_prep_w<<<blocks(256 * 2304), 256, 0, stream>>>(oc2_w, wh_oc2, 256, 2304, 2304);
    k_prep_w<<<blocks(108 * 2304), 256, 0, stream>>>(off_w, wh_off, 108, 2304, 2304);
    k_prep_w<<<blocks(256 * 2304), 256, 0, stream>>>(dcn_w, wh_dcn, 256, 2304, 2304);
    k_prep_w<<<blocks(256 * 4608), 256, 0, stream>>>(rb_c1_w, wh_rc1, 256, 4608, 4608);
    k_prep_w<<<blocks(256 * 512), 256, 0, stream>>>(rb_sc_w, wh_rsc, 256, 512, 512);
    k_prep_w<<<blocks(256 * 2304), 256, 0, stream>>>(rb_c2_w, wh_rc2, 256, 2304, 2304);
    k_prep_w<<<blocks(3 * 2336), 256, 0, stream>>>(out_w, wh_out, 3, 2331, 2336);

    k_vq_argmin<<<512, 32, 0, stream>>>(ze, cbh, cn, idx);
    k_gather_hq_loss<<<8192, 256, 0, stream>>>(ze, cb, idx, hq, loss);

    // offset branch
    k_conv_wmma<1><<<256, cblk, 0, stream>>>(prior, 256, hq, 256, wh_c1, oc1_c1_b, nullptr, t0, 256, 0, 512);
    k_gn_silu<<<256, 256, 0, stream>>>(t0, 256, nullptr, 0, oc1_gn1_g, oc1_gn1_b, t0, 1);
    k_dw7<<<8192, 256, 0, stream>>>(t0, oc1_dw_w, oc1_dw_b, t1);
    k_gn_silu<<<256, 256, 0, stream>>>(t1, 256, nullptr, 0, oc1_gn2_g, oc1_gn2_b, t1, 1);
    k_conv_wmma<1><<<256, cblk, 0, stream>>>(t1, 256, nullptr, 0, wh_c2, oc1_c2_b, nullptr, t0, 256, 0, 256);
    k_conv_wmma<3><<<256, cblk, 0, stream>>>(t0, 256, nullptr, 0, wh_oc2, oc2_b, nullptr, t1, 256, 1, 2304);
    k_gn_silu<<<256, 256, 0, stream>>>(t1, 256, nullptr, 0, oc2_gn_g, oc2_gn_b, t1, 1);
    k_conv_wmma<3><<<256, cblk, 0, stream>>>(t1, 256, nullptr, 0, wh_off, off_b, nullptr, om, 108, 1, 2304);

    // deformable conv
    k_deform_wmma<<<256, cblk, 0, stream>>>(hq, om, wh_dcn, dcn_b, warp);

    // residual block on concat(hq, warp)
    k_gn_silu<<<256, 256, 0, stream>>>(hq, 256, warp, 256, rb_gn1_g, rb_gn1_b, gnb, 1);
    k_conv_wmma<3><<<256, cblk, 0, stream>>>(gnb, 512, nullptr, 0, wh_rc1, rb_c1_b, nullptr, t0, 256, 1, 4608);
    k_gn_silu<<<256, 256, 0, stream>>>(t0, 256, nullptr, 0, rb_gn2_g, rb_gn2_b, t0, 1);
    k_conv_wmma<1><<<256, cblk, 0, stream>>>(hq, 256, warp, 256, wh_rsc, rb_sc_b, nullptr, xb, 256, 0, 512);
    k_conv_wmma<3><<<256, cblk, 0, stream>>>(t0, 256, nullptr, 0, wh_rc2, rb_c2_b, xb, t1, 256, 1, 2304);

    // output head
    k_gn_silu<<<256, 256, 0, stream>>>(t1, 256, nullptr, 0, out_gn_g, out_gn_b, t1, 1);
    k_conv_wmma<3><<<256, cblk, 0, stream>>>(t1, 256, xlq, 3, wh_out, out_b, nullptr, img, 3, 1, 2336);

    k_finalize_loss<<<1, 32, 0, stream>>>(loss, lossout);
}
